// AdaptiveLoss_77000173683136
// MI455X (gfx1250) — matrix operand order repaired and checked by reading, not compiled, not run on hardware
//
#include <hip/hip_runtime.h>
#include <hip/hip_bf16.h>
#include <stdint.h>

// ---------------------------------------------------------------------------
// CDNA5 / gfx1250 wave32 WMMA implementation of the adaptive-softmax loss.
// Round 3: kill the register spills -- no manual double-buffering (rely on
// wave-level latency hiding), uniform-base + 32-bit offsets for all fragment
// loads (GVS addressing, no per-lane 64-bit pointer arithmetic).
// ---------------------------------------------------------------------------

typedef __bf16 bf16;
typedef __attribute__((ext_vector_type(16))) __bf16        v16bf;
typedef __attribute__((ext_vector_type(8)))  float         v8f;
typedef __attribute__((ext_vector_type(4)))  unsigned int  u32x4;

#define WMMA_BF16(A_, B_, C_) \
  __builtin_amdgcn_wmma_f32_16x16x32_bf16(false, (A_), false, (B_), (short)0, (C_), false, false)

// ---- A fragment (16x32 bf16, ISA 7.12.2): two 16B chunks, 32B apart --------
// per-lane element offset = (row)*lda + (lane<16 ? 0 : 8) + k0
__device__ __forceinline__ v16bf load_a_frag(const bf16* __restrict__ base, int off) {
  union { u32x4 u[2]; v16bf v; } r;
  r.u[0] = *(const u32x4*)(base + off);        // K = bk .. bk+7
  r.u[1] = *(const u32x4*)(base + off + 16);   // K = bk+16 .. bk+23
  return r.v;
}

// ---- B fragment (32x16 from bf16 W[N,K] row-major): two 16B chunks, 32B apart
// per-lane element offset = col*K + (lane<16 ? 0 : 8) + k0
__device__ __forceinline__ v16bf load_b_frag(const bf16* __restrict__ base, int off) {
  union { u32x4 u[2]; v16bf v; } r;
  r.u[0] = *(const u32x4*)(base + off);
  r.u[1] = *(const u32x4*)(base + off + 16);
  return r.v;
}

__device__ __forceinline__ float wave_sum32(float p) {
#pragma unroll
  for (int m = 16; m >= 1; m >>= 1) p += __shfl_xor(p, m, 32);
  return p;
}

// ---------------------------------------------------------------------------
// K0: generic fp32 -> bf16 conversion (grid sized exactly: n/4 threads)
// ---------------------------------------------------------------------------
__global__ void cvt_kernel(const float* __restrict__ x, bf16* __restrict__ xb) {
  int i = blockIdx.x * 256 + threadIdx.x;
  float4 f = ((const float4*)x)[i];
  bf16* p = xb + (size_t)i * 4;
  p[0] = (bf16)f.x; p[1] = (bf16)f.y; p[2] = (bf16)f.z; p[3] = (bf16)f.w;
}

// ---------------------------------------------------------------------------
// K2: h[1024,960] = x @ projcat^T  (WMMA bf16, fp32 acc), projcat = pb[960,1024]
// stores fp32 h (for exact target dots) and bf16 h (for cluster normalizers)
// ---------------------------------------------------------------------------
__global__ void proj_gemm_kernel(const bf16* __restrict__ xb,
                                 const bf16* __restrict__ pb,
                                 float* __restrict__ hf, bf16* __restrict__ hb) {
  const int rb0  = blockIdx.x * 32;
  const int c0   = blockIdx.y * 512;
  const int lane = threadIdx.x & 31;
  const int wave = threadIdx.x >> 5;
  const int nb   = c0 + wave * 64;
  const int lo16 = (lane < 16);

  // 32-bit element offsets (uniform base + divergent offset -> GVS addressing)
  const int aoff0 = (rb0 + (lane & 15)) * 1024 + (lo16 ? 0 : 8);
  const int aoff1 = aoff0 + 16 * 1024;
  int boff[4];
#pragma unroll
  for (int i = 0; i < 4; ++i) {
    int n  = nb + i * 16 + (lane & 15);
    int nc = (n < 960) ? n : 959;              // clamp; masked at store
    boff[i] = nc * 1024 + (lo16 ? 0 : 8);
  }

  v8f acc[2][4];
  v8f z = {0.f,0.f,0.f,0.f,0.f,0.f,0.f,0.f};
#pragma unroll
  for (int j = 0; j < 2; ++j)
#pragma unroll
    for (int i = 0; i < 4; ++i) acc[j][i] = z;

  for (int k0 = 0; k0 < 1024; k0 += 32) {
    v16bf a0 = load_a_frag(xb, aoff0 + k0);
    v16bf a1 = load_a_frag(xb, aoff1 + k0);
    v16bf b0 = load_b_frag(pb, boff[0] + k0);
    v16bf b1 = load_b_frag(pb, boff[1] + k0);
    v16bf b2 = load_b_frag(pb, boff[2] + k0);
    v16bf b3 = load_b_frag(pb, boff[3] + k0);
    acc[0][0] = WMMA_BF16(a0, b0, acc[0][0]);
    acc[1][0] = WMMA_BF16(a1, b0, acc[1][0]);
    acc[0][1] = WMMA_BF16(a0, b1, acc[0][1]);
    acc[1][1] = WMMA_BF16(a1, b1, acc[1][1]);
    acc[0][2] = WMMA_BF16(a0, b2, acc[0][2]);
    acc[1][2] = WMMA_BF16(a1, b2, acc[1][2]);
    acc[0][3] = WMMA_BF16(a0, b3, acc[0][3]);
    acc[1][3] = WMMA_BF16(a1, b3, acc[1][3]);
  }

  const int rlo = lo16 ? 0 : 8;   // C layout: VGPR r -> M=r (+8 for hi lanes)
#pragma unroll
  for (int i = 0; i < 4; ++i) {
    int n = nb + i * 16 + (lane & 15);
    if (n < 960) {
#pragma unroll
      for (int j = 0; j < 2; ++j)
#pragma unroll
        for (int r = 0; r < 8; ++r) {
          int row = rb0 + j * 16 + rlo + r;
          float v = acc[j][i][r];
          hf[(size_t)row * 960 + n] = v;
          hb[(size_t)row * 960 + n] = (bf16)v;
        }
    }
  }
}

// ---------------------------------------------------------------------------
// K3: fused logits + exp + per-row partial sums for one softmax group.
// Workgroup tile: 32 rows x 512 cols (8 waves x [2x4] 16x16 WMMA tiles).
// Deterministic: per-wave row sums -> LDS -> fixed-order reduce by wave 0.
// ---------------------------------------------------------------------------
__global__ void sumexp_gemm_kernel(const bf16* __restrict__ A, int lda, int K,
                                   const bf16* __restrict__ W, int Ng,
                                   float* __restrict__ partials) {
  const int rb0   = blockIdx.x * 32;
  const int chunk = blockIdx.y;
  const int lane  = threadIdx.x & 31;
  const int wave  = threadIdx.x >> 5;
  const int nb    = chunk * 512 + wave * 64;
  const int lo16  = (lane < 16);

  const int aoff0 = (rb0 + (lane & 15)) * lda + (lo16 ? 0 : 8);
  const int aoff1 = aoff0 + 16 * lda;
  int boff[4];
#pragma unroll
  for (int i = 0; i < 4; ++i) {
    int n  = nb + i * 16 + (lane & 15);
    int nc = (n < Ng) ? n : (Ng - 1);   // clamp OOB reads; masked before exp
    boff[i] = nc * K + (lo16 ? 0 : 8);
  }

  v8f acc[2][4];
  v8f z = {0.f,0.f,0.f,0.f,0.f,0.f,0.f,0.f};
#pragma unroll
  for (int j = 0; j < 2; ++j)
#pragma unroll
    for (int i = 0; i < 4; ++i) acc[j][i] = z;

  for (int k0 = 0; k0 < K; k0 += 32) {
    v16bf a0 = load_a_frag(A, aoff0 + k0);
    v16bf a1 = load_a_frag(A, aoff1 + k0);
    v16bf b0 = load_b_frag(W, boff[0] + k0);
    v16bf b1 = load_b_frag(W, boff[1] + k0);
    v16bf b2 = load_b_frag(W, boff[2] + k0);
    v16bf b3 = load_b_frag(W, boff[3] + k0);
    acc[0][0] = WMMA_BF16(a0, b0, acc[0][0]);
    acc[1][0] = WMMA_BF16(a1, b0, acc[1][0]);
    acc[0][1] = WMMA_BF16(a0, b1, acc[0][1]);
    acc[1][1] = WMMA_BF16(a1, b1, acc[1][1]);
    acc[0][2] = WMMA_BF16(a0, b2, acc[0][2]);
    acc[1][2] = WMMA_BF16(a1, b2, acc[1][2]);
    acc[0][3] = WMMA_BF16(a0, b3, acc[0][3]);
    acc[1][3] = WMMA_BF16(a1, b3, acc[1][3]);
  }

  // exp + column mask + per-lane row partials (logits are O(1): no max shift)
  float rs[16];
#pragma unroll
  for (int e = 0; e < 16; ++e) rs[e] = 0.f;
#pragma unroll
  for (int i = 0; i < 4; ++i) {
    int n = nb + i * 16 + (lane & 15);
    if (n < Ng) {
#pragma unroll
      for (int j = 0; j < 2; ++j)
#pragma unroll
        for (int r = 0; r < 8; ++r)
          rs[j * 8 + r] += __expf(acc[j][i][r]);
    }
  }
  // reduce across the 16 lanes holding the same row (xor masks stay in-half)
#pragma unroll
  for (int m = 1; m <= 8; m <<= 1)
#pragma unroll
    for (int e = 0; e < 16; ++e) rs[e] += __shfl_xor(rs[e], m, 32);

  __shared__ float wsum[8][32];
  if ((lane & 15) == 0) {
    int rbase = lo16 ? 0 : 8;
#pragma unroll
    for (int j = 0; j < 2; ++j)
#pragma unroll
      for (int r = 0; r < 8; ++r)
        wsum[wave][j * 16 + rbase + r] = rs[j * 8 + r];
  }
  __syncthreads();
  if (threadIdx.x < 32) {
    float t = 0.f;
#pragma unroll
    for (int w = 0; w < 8; ++w) t += wsum[w][threadIdx.x];   // fixed order
    partials[(size_t)chunk * 1024 + rb0 + threadIdx.x] = t;
  }
}

// ---------------------------------------------------------------------------
// K4: per-row LSE per group: lse[g*1024+row] = log(sum_chunks partials)
// ---------------------------------------------------------------------------
__global__ void lse_kernel(const float* __restrict__ partials, float* __restrict__ lse) {
  int idx = blockIdx.x * 256 + threadIdx.x;       // 5*1024 values
  if (idx >= 5 * 1024) return;
  int g = idx >> 10, row = idx & 1023;
  const int nchunk[5] = {20, 20, 40, 79, 40};
  const float* p = partials + (size_t)g * 80 * 1024 + row;
  float s = 0.f;
  for (int c = 0; c < nchunk[g]; ++c) s += p[(size_t)c * 1024];
  lse[idx] = logf(s);
}

// ---------------------------------------------------------------------------
// K5: per-row gather loss. One block per feature row b; exact fp32 dots for
// the target logit and the 4 head-tail logits. Wave-uniform targets.
// vocab: head [0,10k) | c0 [10k,20k) | c1 [20k,40k) | c2 [40k,80k) | c3 [80k,100k)
// ---------------------------------------------------------------------------
__global__ void loss_kernel(const float* __restrict__ x,  const float* __restrict__ hw,
                            const float* __restrict__ o0, const float* __restrict__ o1,
                            const float* __restrict__ o2, const float* __restrict__ o3,
                            const float* __restrict__ hf, const float* __restrict__ lse,
                            const float* __restrict__ discard,
                            const int* __restrict__ targets,
                            const unsigned char* __restrict__ mask,
                            float* __restrict__ per_sample) {
  const int b    = blockIdx.x;
  const int tid  = threadIdx.x;
  const int lane = tid & 31;
  const int wave = tid >> 5;

  __shared__ float xs[1024];
  __shared__ float hs[960];
  __shared__ float tail[4];
  __shared__ float ls[5];
  __shared__ float wnum[8], wden[8];

  for (int i = tid; i < 1024; i += 256) xs[i] = x[(size_t)b * 1024 + i];
  for (int i = tid; i < 960;  i += 256) hs[i] = hf[(size_t)b * 960 + i];
  if (tid < 5) ls[tid] = lse[tid * 1024 + b];
  __syncthreads();

  if (wave < 4) {  // exact fp32 tail logits: dot(x, head_weight[10000+i])
    const float* wr = hw + (size_t)(10000 + wave) * 1024;
    float p = 0.f;
    for (int k = lane; k < 1024; k += 32) p += xs[k] * wr[k];
    p = wave_sum32(p);
    if (lane == 0) tail[wave] = p;
  }
  __syncthreads();

  float num = 0.f, den = 0.f;
  for (int s = 0; s < 16; ++s) {
    int t = wave + (s << 3);                       // 8 waves x 16 = 128 targets
    int v = targets[b * 128 + t];
    float w = (1.0f - discard[v]) * (mask[b * 128 + t] ? 1.0f : 0.0f);
    float lp;
    if (v < 10000) {
      const float* wr = hw + (size_t)v * 1024;
      float p = 0.f;
      for (int k = lane; k < 1024; k += 32) p += xs[k] * wr[k];
      p = wave_sum32(p);
      lp = p - ls[0];
    } else {
      int g, off, hofs, Kc; const float* base;
      if      (v < 20000) { g = 0; base = o0; off = v - 10000; hofs = 0;   Kc = 512; }
      else if (v < 40000) { g = 1; base = o1; off = v - 20000; hofs = 512; Kc = 256; }
      else if (v < 80000) { g = 2; base = o2; off = v - 40000; hofs = 768; Kc = 128; }
      else                { g = 3; base = o3; off = v - 80000; hofs = 896; Kc = 64;  }
      const float* wr = base + (size_t)off * Kc;
      float p = 0.f;
      for (int k = lane; k < Kc; k += 32) p += hs[hofs + k] * wr[k];
      p = wave_sum32(p);
      lp = (p - ls[1 + g]) + (tail[g] - ls[0]);
    }
    num += -lp * w;
    den += w;
  }
  if (lane == 0) { wnum[wave] = num; wden[wave] = den; }
  __syncthreads();
  if (tid == 0) {
    float n = 0.f, d = 0.f;
#pragma unroll
    for (int w = 0; w < 8; ++w) { n += wnum[w]; d += wden[w]; }  // fixed order
    per_sample[b] = n / d;
  }
}

// ---------------------------------------------------------------------------
// K6: deterministic final reduction to the scalar loss.
// ---------------------------------------------------------------------------
__global__ void final_kernel(const float* __restrict__ per_sample, float* __restrict__ out) {
  __shared__ float buf[256];
  int tid = threadIdx.x;
  float s = 0.f;
  for (int i = tid; i < 1024; i += 256) s += per_sample[i];
  buf[tid] = s;
  __syncthreads();
  for (int w = 128; w > 0; w >>= 1) {
    if (tid < w) buf[tid] += buf[tid + w];
    __syncthreads();
  }
  if (tid == 0) out[0] = buf[0] / (1024.0f + 1e-5f);
}

// ---------------------------------------------------------------------------
// Launch
// ---------------------------------------------------------------------------
extern "C" void kernel_launch(void* const* d_in, const int* in_sizes, int n_in,
                              void* d_out, int out_size, void* d_ws, size_t ws_size,
                              hipStream_t stream) {
  (void)in_sizes; (void)n_in; (void)out_size; (void)ws_size;

  const float* features = (const float*)d_in[0];   // [1024,1024]
  const float* head_w   = (const float*)d_in[1];   // [10004,1024]
  const float* proj0    = (const float*)d_in[2];   // [512,1024]
  const float* out0     = (const float*)d_in[3];   // [10000,512]
  const float* proj1    = (const float*)d_in[4];   // [256,1024]
  const float* out1     = (const float*)d_in[5];   // [20000,256]
  const float* proj2    = (const float*)d_in[6];   // [128,1024]
  const float* out2     = (const float*)d_in[7];   // [40000,128]
  const float* proj3    = (const float*)d_in[8];   // [64,1024]
  const float* out3     = (const float*)d_in[9];   // [20000,64]
  const float* discard  = (const float*)d_in[10];  // [100000]
  const int*   targets  = (const int*)d_in[11];    // [1024,128] (integer -> int per harness)
  const unsigned char* mask = (const unsigned char*)d_in[12]; // bool [1024,128]
  float* out = (float*)d_out;

  // workspace layout (bytes, 256-aligned), ~65.4 MB total
  char*  ws       = (char*)d_ws;
  bf16*  xb       = (bf16*)(ws + 0);           // 1024*1024  bf16 =  2,097,152
  float* hf       = (float*)(ws + 2097152);    // 1024*960   fp32 =  3,932,160
  bf16*  hb       = (bf16*) (ws + 6029312);    // 1024*960   bf16 =  1,966,080
  float* partials = (float*)(ws + 7995392);    // 5*80*1024  fp32 =  1,638,400
  float* lse      = (float*)(ws + 9633792);    // 5*1024     fp32 =     20,480
  float* per_s    = (float*)(ws + 9654272);    // 1024       fp32 =      4,096
  bf16*  pb       = (bf16*)(ws + 9658368);     // 960*1024   bf16 =  1,966,080
  bf16*  hwb      = (bf16*)(ws + 11624448);    // 10004*1024 bf16 = 20,488,192
  bf16*  ob0      = (bf16*)(ws + 32112640);    // 10000*512  bf16 = 10,240,000
  bf16*  ob1      = (bf16*)(ws + 42352640);    // 20000*256  bf16 = 10,240,000
  bf16*  ob2      = (bf16*)(ws + 52592640);    // 40000*128  bf16 = 10,240,000
  bf16*  ob3      = (bf16*)(ws + 62832640);    // 20000*64   bf16 =  2,560,000

  // ---- fp32 -> bf16 conversions (grids are exact: count/4/256 integral) ----
  cvt_kernel<<<1024,  256, 0, stream>>>(features, xb);
  cvt_kernel<<<512,   256, 0, stream>>>(proj0, pb + 0 * 1024);        // rows   0..511
  cvt_kernel<<<256,   256, 0, stream>>>(proj1, pb + 512 * 1024);      // rows 512..767
  cvt_kernel<<<128,   256, 0, stream>>>(proj2, pb + 768 * 1024);      // rows 768..895
  cvt_kernel<<<64,    256, 0, stream>>>(proj3, pb + 896 * 1024);      // rows 896..959
  cvt_kernel<<<10004, 256, 0, stream>>>(head_w, hwb);
  cvt_kernel<<<5000,  256, 0, stream>>>(out0, ob0);
  cvt_kernel<<<5000,  256, 0, stream>>>(out1, ob1);
  cvt_kernel<<<5000,  256, 0, stream>>>(out2, ob2);
  cvt_kernel<<<1250,  256, 0, stream>>>(out3, ob3);

  // ---- projections: h = x @ projcat^T ----
  proj_gemm_kernel<<<dim3(32, 2), 256, 0, stream>>>(xb, pb, hf, hb);

  // ---- normalizer GEMMs: (A, lda, K, Wbf16, Ng, partials+g*80*1024) --------
  sumexp_gemm_kernel<<<dim3(32, 20), 256, 0, stream>>>(xb,       1024, 1024, hwb, 10004, partials + 0 * 80 * 1024);
  sumexp_gemm_kernel<<<dim3(32, 20), 256, 0, stream>>>(hb + 0,    960,  512, ob0, 10000, partials + 1 * 80 * 1024);
  sumexp_gemm_kernel<<<dim3(32, 40), 256, 0, stream>>>(hb + 512,  960,  256, ob1, 20000, partials + 2 * 80 * 1024);
  sumexp_gemm_kernel<<<dim3(32, 79), 256, 0, stream>>>(hb + 768,  960,  128, ob2, 40000, partials + 3 * 80 * 1024);
  sumexp_gemm_kernel<<<dim3(32, 40), 256, 0, stream>>>(hb + 896,  960,   64, ob3, 20000, partials + 4 * 80 * 1024);

  lse_kernel<<<20, 256, 0, stream>>>(partials, lse);

  loss_kernel<<<1024, 256, 0, stream>>>(features, head_w, out0, out1, out2, out3,
                                        hf, lse, discard, targets, mask, per_s);

  final_kernel<<<1, 256, 0, stream>>>(per_s, out);
}
